// GCNN_84189948936394
// MI455X (gfx1250) — matrix operand order repaired and checked by reading, not compiled, not run on hardware
//
#include <hip/hip_runtime.h>
#include <hip/hip_bf16.h>

typedef __attribute__((ext_vector_type(16))) __bf16 v16bf;
typedef __attribute__((ext_vector_type(8)))  float  v8f;

#define T_N   209
#define R_N   43472
#define D_N   512
#define OBJN  192
#define RELM  36864   // 192*192

__device__ __forceinline__ unsigned short f2bf(float x) {
  unsigned int u = __float_as_uint(x);
  u += 0x7FFFu + ((u >> 16) & 1u);          // round-to-nearest-even
  return (unsigned short)(u >> 16);
}

// ---------------------------------------------------------------------------
// One-time weight transpose + bf16 convert: W[k][n] (512x512 f32) -> Wt[n][k]
// ---------------------------------------------------------------------------
__global__ void k_transpose_bf16(const float* __restrict__ W,
                                 unsigned short* __restrict__ Wt) {
  int n = blockIdx.x;
  for (int k = threadIdx.x; k < D_N; k += blockDim.x)
    Wt[n * D_N + k] = f2bf(W[k * D_N + n]);
}

// ---------------------------------------------------------------------------
// Node MLP layer 1: h = relu(x @ w1 + b1), per-row type select (obj/wall/floor)
// ---------------------------------------------------------------------------
__global__ void k_node_l1(const float* __restrict__ objf, const float* __restrict__ wallf,
                          const float* __restrict__ floorf,
                          const float* __restrict__ ow1, const float* __restrict__ ob1,
                          const float* __restrict__ ww1, const float* __restrict__ wb1,
                          const float* __restrict__ fw1, const float* __restrict__ fb1,
                          float* __restrict__ h) {
  int id = blockIdx.x * blockDim.x + threadIdx.x;
  if (id >= T_N * D_N) return;
  int r = id >> 9, c = id & 511;
  const float *x, *w, *b; int din;
  if (r < OBJN)      { x = objf  + r * 42;          w = ow1; b = ob1; din = 42; }
  else if (r < 208)  { x = wallf + (r - OBJN) * 15; w = ww1; b = wb1; din = 15; }
  else               { x = floorf;                  w = fw1; b = fb1; din = 16; }
  float acc = b[c];
  for (int i = 0; i < din; ++i) acc += x[i] * w[i * D_N + c];
  h[id] = acc > 0.f ? acc : 0.f;
}

// Node MLP layer 2: tot = h @ w2 + b2 (per-row type select). 209x512, K=512.
__global__ void k_node_l2(const float* __restrict__ h,
                          const float* __restrict__ ow2, const float* __restrict__ ob2,
                          const float* __restrict__ ww2, const float* __restrict__ wb2,
                          const float* __restrict__ fw2, const float* __restrict__ fb2,
                          float* __restrict__ tot) {
  int id = blockIdx.x * blockDim.x + threadIdx.x;
  if (id >= T_N * D_N) return;
  int r = id >> 9, c = id & 511;
  const float *w, *b;
  if (r < OBJN)     { w = ow2; b = ob2; }
  else if (r < 208) { w = ww2; b = wb2; }
  else              { w = fw2; b = fb2; }
  const float* hr = h + r * D_N;
  float acc = b[c];
  for (int k = 0; k < D_N; ++k) acc += hr[k] * w[k * D_N + c];
  tot[id] = acc;
}

// ---------------------------------------------------------------------------
// Relation MLP layer 1: K=2, done elementwise. h[r][c] = relu(x0*w+x1*w+b)
// ---------------------------------------------------------------------------
__global__ void k_rel_l1(const float* __restrict__ xf, const float* __restrict__ w1,
                         const float* __restrict__ b1, float* __restrict__ h) {
  int id = blockIdx.x * blockDim.x + threadIdx.x;
  if (id >= RELM * D_N) return;
  int r = id >> 9, c = id & 511;
  float acc = xf[r * 2] * w1[c] + xf[r * 2 + 1] * w1[D_N + c] + b1[c];
  h[id] = acc > 0.f ? acc : 0.f;
}

// Fill the padded (non-object) relation rows with the constant 0.001
__global__ void k_rel_fill(float* __restrict__ rel) {
  int r = blockIdx.x;
  int s = r / 208;
  int k = r - s * 208;
  int o = k + (k >= s ? 1 : 0);
  if (s < OBJN && o < OBJN) return;
  rel[(long)r * D_N + threadIdx.x] = 0.001f;
}

// ---------------------------------------------------------------------------
// WMMA GEMM: C = A(virtual rows, f32, Mx512) @ Wt^T (bf16 pre-transposed [n][k])
// MODE 0: subject-grouped reduce epilogue -> atomicAdd into out[209][512]
// MODE 1: object-grouped   reduce epilogue (gathered A rows) -> out[209][512]
// MODE 2: relation-embed layer2: store rows scattered into rel[43472][512]
// Block = 256 threads = 8 waves; TWO 16-row M-tiles per block (B fragments are
// reused by two wmmas -> wmma : global_load_b128 = 1 : 1); wave covers 64 N.
// ---------------------------------------------------------------------------
__device__ __forceinline__ int map_row(int MODE, int mtile, int p) {
  if (MODE == 1) {
    int t  = mtile / 13;               // object id (13 tiles of 16 = 208 subjects)
    int j  = mtile - t * 13;
    int sp = j * 16 + p;               // 0..207
    int s  = sp + (sp >= t ? 1 : 0);   // skip s == t
    return s * 208 + (s < t ? t - 1 : t);
  }
  return mtile * 16 + p;               // identity (subject blocks are contiguous)
}

template <int MODE>
__device__ __forceinline__ void epilogue(int mtile, const v8f* acc,
                                         int n0, int idx, int half,
                                         const float* __restrict__ bias,
                                         float* __restrict__ out) {
  if (MODE == 2) {
    // Scatter rows of emb_rel into rel via off-diagonal index map
#pragma unroll
    for (int t = 0; t < 4; ++t) {
      float bv = bias[n0 + t * 16 + idx];
#pragma unroll
      for (int j = 0; j < 8; ++j) {
        int e = mtile * 16 + j + 8 * half;    // row of 192x192 block
        int s = e / OBJN;
        int o = e - s * OBJN;
        if (s != o) {
          int r = s * 208 + o - (o > s ? 1 : 0);
          out[(long)r * D_N + n0 + t * 16 + idx] = acc[t][j] + bv;
        }
      }
    }
  } else {
    // relu + column-sum over the 16 tile rows (all same subject/object bucket),
    // cross-half combine via wave32 xor-16 shuffle, one atomic per column.
    int bucket = mtile / 13;
#pragma unroll
    for (int t = 0; t < 4; ++t) {
      float bv = bias[n0 + t * 16 + idx];
      float s = 0.f;
#pragma unroll
      for (int j = 0; j < 8; ++j) {
        float v = acc[t][j] + bv;
        s += v > 0.f ? v : 0.f;
      }
      s += __shfl_xor(s, 16, 32);
      if (half == 0)
        atomicAdd(out + bucket * D_N + n0 + t * 16 + idx, s);
    }
  }
}

template <int MODE>
__global__ void __launch_bounds__(256)
k_wmma_gemm(const float* __restrict__ A, const unsigned short* __restrict__ Wt,
            const float* __restrict__ bias, float* __restrict__ out, int mtiles) {
  __shared__ __align__(16) unsigned short Abuf[32][520];  // +8 pad: conflict-free b128
  const int mt0 = 2 * blockIdx.x;
  const int mt1 = mt0 + 1;
  const int tid = threadIdx.x;

  // Stage two A tiles (32 x 512) f32 -> bf16 into LDS
  {
    int p  = tid >> 3;                 // staged row 0..31
    int c0 = (tid & 7) * 64;           // 64 cols per thread
    int mt = mt0 + (p >> 4);
    if (mt >= mtiles) mt = mtiles - 1; // clamp: safe reads for the odd tail block
    int row = map_row(MODE, mt, p & 15);
    const float* src = A + (long)row * D_N + c0;
#pragma unroll
    for (int i = 0; i < 64; i += 4) {
      float4 f = *(const float4*)(src + i);
      Abuf[p][c0 + i + 0] = f2bf(f.x);
      Abuf[p][c0 + i + 1] = f2bf(f.y);
      Abuf[p][c0 + i + 2] = f2bf(f.z);
      Abuf[p][c0 + i + 3] = f2bf(f.w);
    }
  }
  __syncthreads();

  const int wave = tid >> 5;
  const int lane = tid & 31;
  const int half = lane >> 4;          // wave32: two 16-lane halves
  const int idx  = lane & 15;
  const int n0   = wave * 64;          // 4 N-tiles per wave

  v8f acc0[4] = {};
  v8f acc1[4] = {};
  for (int k0 = 0; k0 < D_N; k0 += 32) {
    // A fragments 16x32: lane idx holds row m=idx; half 0 -> K {0..7,16..23},
    // half 1 -> K {8..15,24..31} (ISA 7.12.2 16-bit A layout)
    union { uint4 u[2]; v16bf v; } a0, a1;
    const unsigned short* ap0 = &Abuf[idx][k0 + half * 8];
    const unsigned short* ap1 = &Abuf[16 + idx][k0 + half * 8];
    a0.u[0] = *(const uint4*)ap0;
    a0.u[1] = *(const uint4*)(ap0 + 16);
    a1.u[0] = *(const uint4*)ap1;
    a1.u[1] = *(const uint4*)(ap1 + 16);
#pragma unroll
    for (int t = 0; t < 4; ++t) {
      // B fragment 32x16: lane idx = column n; half 0 -> K {0..7,16..23}, half 1 -> K {8..15,24..31}
      union { uint4 u[2]; v16bf v; } b;
      const unsigned short* bp = Wt + (n0 + t * 16 + idx) * D_N + k0 + half * 8;
      b.u[0] = *(const uint4*)bp;
      b.u[1] = *(const uint4*)(bp + 16);
      acc0[t] = __builtin_amdgcn_wmma_f32_16x16x32_bf16(false, a0.v, false, b.v,
                                                        (short)0, acc0[t], false, false);
      acc1[t] = __builtin_amdgcn_wmma_f32_16x16x32_bf16(false, a1.v, false, b.v,
                                                        (short)0, acc1[t], false, false);
    }
  }

  epilogue<MODE>(mt0, acc0, n0, idx, half, bias, out);
  if (mt1 < mtiles)
    epilogue<MODE>(mt1, acc1, n0, idx, half, bias, out);
}

// ---------------------------------------------------------------------------
// fc2/fc3/fc4 = relu(tot @ collect_w[2/3/4] + b) : 3 x 209x512, K=512 (tiny)
// ---------------------------------------------------------------------------
__global__ void k_fc_tot(const float* __restrict__ tot, const float* __restrict__ cw,
                         const float* __restrict__ cb, float* __restrict__ fc2,
                         float* __restrict__ fc3, float* __restrict__ fc4) {
  int which = blockIdx.y;              // 0->W2, 1->W3, 2->W4
  int id = blockIdx.x * blockDim.x + threadIdx.x;
  if (id >= T_N * D_N) return;
  int r = id >> 9, c = id & 511;
  const float* w = cw + (2 + which) * D_N * D_N;
  float acc = cb[(2 + which) * D_N + c];
  const float* tr = tot + r * D_N;
  for (int k = 0; k < D_N; ++k) acc += tr[k] * w[k * D_N + c];
  acc = acc > 0.f ? acc : 0.f;
  float* dst = which == 0 ? fc2 : (which == 1 ? fc3 : fc4);
  dst[id] = acc;
}

// s_obj[c] = mean over 209 rows of fc4, incl. 1e-7 denominator fudge
__global__ void k_sobj(const float* __restrict__ fc4, float* __restrict__ sobj) {
  int c = threadIdx.x;
  float s = 0.f;
  for (int t = 0; t < T_N; ++t) s += fc4[t * D_N + c];
  sobj[c] = s / 209.0000001f;
}

// rel[r] += (fc2[subj(r)] + fc3[obj(r)]) / 2 / (1+1e-7)
__global__ void k_rel_update(float* __restrict__ rel, const float* __restrict__ fc2,
                             const float* __restrict__ fc3) {
  int r = blockIdx.x;
  int c = threadIdx.x;
  int s = r / 208;
  int k = r - s * 208;
  int o = k + (k >= s ? 1 : 0);
  const float inv1 = 1.0f / (1.0f + 1e-7f);
  rel[(long)r * D_N + c] += 0.5f * inv1 * (fc2[s * D_N + c] + fc3[o * D_N + c]);
}

// tot[t] += (s_obj + s_rs[t]/208.0000001 + s_ro[t]/208.0000001) / 3
__global__ void k_tot_update(float* __restrict__ tot, const float* __restrict__ sobj,
                             const float* __restrict__ srs, const float* __restrict__ sro) {
  int id = blockIdx.x * blockDim.x + threadIdx.x;
  if (id >= T_N * D_N) return;
  int c = id & 511;
  const float inv208 = 1.0f / 208.0000001f;
  tot[id] += (sobj[c] + srs[id] * inv208 + sro[id] * inv208) * (1.0f / 3.0f);
}

// ---------------------------------------------------------------------------
// Heads: h = leaky_relu(obj @ w1 + b1, 0.2); out = h @ w2 + b2
// ---------------------------------------------------------------------------
__global__ void k_head_l1(const float* __restrict__ tot, const float* __restrict__ w1,
                          const float* __restrict__ b1, float* __restrict__ h) {
  int id = blockIdx.x * blockDim.x + threadIdx.x;
  if (id >= OBJN * 256) return;
  int r = id >> 8, j = id & 255;
  const float* x = tot + r * D_N;
  float acc = b1[j];
  for (int k = 0; k < D_N; ++k) acc += x[k] * w1[k * 256 + j];
  h[id] = acc >= 0.f ? acc : 0.2f * acc;
}

__global__ void k_head_l2(const float* __restrict__ h, const float* __restrict__ w2,
                          const float* __restrict__ b2, float* __restrict__ out, int hd) {
  int id = blockIdx.x * blockDim.x + threadIdx.x;
  if (id >= OBJN * 3) return;
  int r = id / 3, c = id - r * 3;
  const float* x = h + r * 256;
  float acc = b2[c];
  for (int j = 0; j < 256; ++j) acc += x[j] * w2[j * 3 + c];
  out[r * 9 + hd * 3 + c] = acc;
}

// ---------------------------------------------------------------------------
extern "C" void kernel_launch(void* const* d_in, const int* in_sizes, int n_in,
                              void* d_out, int out_size, void* d_ws, size_t ws_size,
                              hipStream_t stream) {
  (void)in_sizes; (void)n_in; (void)out_size; (void)ws_size;

  const float* cat_obj   = (const float*)d_in[0];
  const float* cat_wall  = (const float*)d_in[1];
  const float* cat_floor = (const float*)d_in[2];
  const float* cat_rel   = (const float*)d_in[3];
  const float* obj_w1   = (const float*)d_in[4];
  const float* obj_b1   = (const float*)d_in[5];
  const float* obj_w2   = (const float*)d_in[6];
  const float* obj_b2   = (const float*)d_in[7];
  const float* wall_w1  = (const float*)d_in[8];
  const float* wall_b1  = (const float*)d_in[9];
  const float* wall_w2  = (const float*)d_in[10];
  const float* wall_b2  = (const float*)d_in[11];
  const float* floor_w1 = (const float*)d_in[12];
  const float* floor_b1 = (const float*)d_in[13];
  const float* floor_w2 = (const float*)d_in[14];
  const float* floor_b2 = (const float*)d_in[15];
  const float* rel_w1   = (const float*)d_in[16];
  const float* rel_b1   = (const float*)d_in[17];
  const float* rel_w2   = (const float*)d_in[18];
  const float* rel_b2   = (const float*)d_in[19];
  const float* collect_w = (const float*)d_in[20];
  const float* collect_b = (const float*)d_in[21];
  float* out = (float*)d_out;

  // Workspace carve-up (peak ~170 MB; rel state L2-resident at 89 MB)
  char* ws = (char*)d_ws;
  size_t off = 0;
  auto take = [&](size_t bytes) -> char* {
    char* p = ws + off;
    off += (bytes + 255) & ~(size_t)255;
    return p;
  };
  float* rel    = (float*)take((size_t)R_N * D_N * 4);
  float* h_rel  = (float*)take((size_t)RELM * D_N * 4);
  unsigned short* wt_rel = (unsigned short*)take((size_t)D_N * D_N * 2);
  unsigned short* wt_c0  = (unsigned short*)take((size_t)D_N * D_N * 2);
  unsigned short* wt_c1  = (unsigned short*)take((size_t)D_N * D_N * 2);
  float* tot   = (float*)take((size_t)T_N * D_N * 4);
  float* hnode = (float*)take((size_t)T_N * D_N * 4);
  float* fc2   = (float*)take((size_t)T_N * D_N * 4);
  float* fc3   = (float*)take((size_t)T_N * D_N * 4);
  float* fc4   = (float*)take((size_t)T_N * D_N * 4);
  float* s_rs  = (float*)take((size_t)T_N * D_N * 4);
  float* s_ro  = (float*)take((size_t)T_N * D_N * 4);
  float* sobj  = (float*)take((size_t)D_N * 4);
  float* hh    = (float*)take((size_t)OBJN * 256 * 4);

  // --- one-time bf16 weight transposes (B operands of the big GEMMs) ---
  k_transpose_bf16<<<D_N, 256, 0, stream>>>(rel_w2, wt_rel);
  k_transpose_bf16<<<D_N, 256, 0, stream>>>(collect_w + 0 * D_N * D_N, wt_c0);
  k_transpose_bf16<<<D_N, 256, 0, stream>>>(collect_w + 1 * D_N * D_N, wt_c1);

  // --- node embeddings -> tot (209x512) ---
  int nodeBlocks = (T_N * D_N + 255) / 256;
  k_node_l1<<<nodeBlocks, 256, 0, stream>>>(cat_obj, cat_wall, cat_floor,
                                            obj_w1, obj_b1, wall_w1, wall_b1,
                                            floor_w1, floor_b1, hnode);
  k_node_l2<<<nodeBlocks, 256, 0, stream>>>(hnode, obj_w2, obj_b2, wall_w2, wall_b2,
                                            floor_w2, floor_b2, tot);

  // --- relation embeddings -> rel (43472x512) ---
  const int MT_REL = RELM / 16;        // 2304 (even)
  const int MT_R   = R_N / 16;         // 2717 (odd)
  k_rel_l1<<<(RELM * D_N + 255) / 256, 256, 0, stream>>>(cat_rel, rel_w1, rel_b1, h_rel);
  k_wmma_gemm<2><<<MT_REL / 2, 256, 0, stream>>>(h_rel, wt_rel, rel_b2, rel, MT_REL);
  k_rel_fill<<<R_N, D_N, 0, stream>>>(rel);

  // --- 4 message-passing steps ---
  for (int step = 0; step < 4; ++step) {
    hipMemsetAsync(s_rs, 0, (size_t)T_N * D_N * 4, stream);
    hipMemsetAsync(s_ro, 0, (size_t)T_N * D_N * 4, stream);
    // subject-grouped: fused relu(rel@W0+b0) + per-subject column sums
    k_wmma_gemm<0><<<(MT_R + 1) / 2, 256, 0, stream>>>(rel, wt_c0,
                                                       collect_b + 0 * D_N, s_rs, MT_R);
    // object-grouped (gathered A rows): fused relu(rel@W1+b1) + per-object sums
    k_wmma_gemm<1><<<(MT_R + 1) / 2, 256, 0, stream>>>(rel, wt_c1,
                                                       collect_b + 1 * D_N, s_ro, MT_R);
    // node-side fc's from OLD tot
    k_fc_tot<<<dim3(nodeBlocks, 3), 256, 0, stream>>>(tot, collect_w, collect_b,
                                                      fc2, fc3, fc4);
    k_sobj<<<1, D_N, 0, stream>>>(fc4, sobj);
    // rel += (fc2[subj]+fc3[obj])/2  (uses old tot; runs after GEMMs read rel)
    k_rel_update<<<R_N, D_N, 0, stream>>>(rel, fc2, fc3);
    // tot += (s_obj + s_rs/208 + s_ro/208)/3
    k_tot_update<<<nodeBlocks, 256, 0, stream>>>(tot, sobj, s_rs, s_ro);
  }

  // --- heads: tr / eu / sc -> out (192x9) ---
  for (int hd = 0; hd < 3; ++hd) {
    const float* w1 = (const float*)d_in[22 + 4 * hd];
    const float* b1 = (const float*)d_in[23 + 4 * hd];
    const float* w2 = (const float*)d_in[24 + 4 * hd];
    const float* b2 = (const float*)d_in[25 + 4 * hd];
    k_head_l1<<<(OBJN * 256 + 255) / 256, 256, 0, stream>>>(tot, w1, b1, hh);
    k_head_l2<<<(OBJN * 3 + 255) / 256, 256, 0, stream>>>(hh, w2, b2, out, hd);
  }
}
